// LSTM_Decoder_5574867550714
// MI455X (gfx1250) — matrix operand order repaired and checked by reading, not compiled, not run on hardware
//
#include <hip/hip_runtime.h>
#include <hip/hip_bf16.h>

// ---------------- problem constants ----------------
#define B_   64
#define IN_  128
#define H_   1536
#define G4_  (4 * H_)        // 6144
#define T_   32
#define KC0  (IN_ + H_)      // 1664 : layer-0 concatenated K  [x | h]
#define NWG  48              // persistent workgroups (two 16-col j-tiles each)
#define NTHR 256             // 8 waves of 32
#define NBAR (3 * T_)        // 3 device barriers per timestep

typedef __bf16 bf16_t;
typedef __attribute__((ext_vector_type(16))) __bf16 v16bf;
typedef __attribute__((ext_vector_type(8)))  float  v8f;

// ---------------- workspace layout (bytes, all 256-aligned) ----------------
#define OFF_WCAT0  ((size_t)0)                                  // bf16 [G4_][KC0]
#define OFF_W1S    (OFF_WCAT0 + (size_t)G4_ * KC0 * 2)          // bf16 [G4_][H_]
#define OFF_WLIN   (OFF_W1S   + (size_t)G4_ * H_  * 2)          // bf16 [IN_][H_]
#define OFF_BIAS0  (OFF_WLIN  + (size_t)IN_ * H_  * 2)          // f32  [G4_]
#define OFF_BIAS1  (OFF_BIAS0 + (size_t)G4_ * 4)                // f32  [G4_]
#define OFF_XH     (OFF_BIAS1 + (size_t)G4_ * 4)                // bf16 [B_][KC0]
#define OFF_H1B    (OFF_XH    + (size_t)B_ * KC0 * 2)           // bf16 [B_][H_]
#define OFF_GATES  (OFF_H1B   + (size_t)B_ * H_  * 2)           // f32  [B_][G4_]
#define OFF_CBUF   (OFF_GATES + (size_t)B_ * G4_ * 4)           // f32  [B_][H_]
#define OFF_ARRIVE (OFF_CBUF  + (size_t)B_ * H_  * 4)           // u32  [NBAR][NWG]

// ---------------- helpers ----------------
__device__ __forceinline__ float sigmoidf_(float x) { return 1.0f / (1.0f + __expf(-x)); }

__device__ __forceinline__ v8f wmma_bf16(v16bf a, v16bf b, v8f c) {
    return __builtin_amdgcn_wmma_f32_16x16x32_bf16(false, a, false, b, (short)0, c, false, false);
}

// A matrix (16x32, MxK) fragment: lane<16 holds row M=lane, K = k0+[0..7] and k0+[16..23];
// lane>=16 holds row M=lane-16, K = k0+[8..15] and k0+[24..31].
__device__ __forceinline__ v16bf load_a_frag(const bf16_t* base, int stride, int m0, int k0, int lane) {
    int m  = m0 + (lane & 15);
    int ka = k0 + ((lane >> 4) << 3);
    const bf16_t* p = base + (size_t)m * stride + ka;
    union { v16bf v; float4 f[2]; } u;
    u.f[0] = *(const float4*)(p);
    u.f[1] = *(const float4*)(p + 16);
    return u.v;
}

// B matrix (32x16, KxN) fragment from row-major [N][K] weights:
// lane<16 -> col N=n0+lane, K = k0+[0..7] and k0+[16..23]; lane>=16 -> K = k0+[8..15] and k0+[24..31].
__device__ __forceinline__ v16bf load_b_frag(const bf16_t* base, int stride, int n0, int k0, int lane) {
    int n  = n0 + (lane & 15);
    int kb = k0 + ((lane >> 4) << 3);
    const bf16_t* p = base + (size_t)n * stride + kb;
    union { v16bf v; float4 f[2]; } u;
    u.f[0] = *(const float4*)(p);
    u.f[1] = *(const float4*)(p + 16);
    return u.v;
}

// device-wide split barrier: one slot per (barrier, WG); slots pre-zeroed by init kernel
__device__ __forceinline__ void grid_barrier(unsigned int* arrive, int bar, int wg, int tid) {
    __syncthreads();
    if (tid == 0) {
        __hip_atomic_store(&arrive[bar * NWG + wg], 1u, __ATOMIC_RELEASE, __HIP_MEMORY_SCOPE_AGENT);
    }
    if (tid < NWG) {
        while (__hip_atomic_load(&arrive[bar * NWG + tid], __ATOMIC_ACQUIRE,
                                 __HIP_MEMORY_SCOPE_AGENT) == 0u) {
            __builtin_amdgcn_s_sleep(1);
        }
    }
    __syncthreads();
}

// ---------------- init: weight conversion / packing + state init ----------------
__global__ void lstm_init_kernel(
    const float* __restrict__ Wih0, const float* __restrict__ Whh0,
    const float* __restrict__ bih0, const float* __restrict__ bhh0,
    const float* __restrict__ Wih1, const float* __restrict__ Whh1,
    const float* __restrict__ bih1, const float* __restrict__ bhh1,
    const float* __restrict__ Wlin,
    const float* __restrict__ x0, const float* __restrict__ h0, const float* __restrict__ c0,
    bf16_t* Wcat0, bf16_t* W1s, bf16_t* Wlinb, float* bias0, float* bias1,
    bf16_t* xh, float* cbuf, unsigned int* arrive)
{
    size_t tid = (size_t)blockIdx.x * blockDim.x + threadIdx.x;
    size_t nth = (size_t)gridDim.x * blockDim.x;

    // Wcat0[n][k] = k<IN ? Wih0[n][k] : Whh0[n][k-IN]   (bf16)
    for (size_t i = tid; i < (size_t)G4_ * KC0; i += nth) {
        int n = (int)(i / KC0), k = (int)(i % KC0);
        float v = (k < IN_) ? Wih0[(size_t)n * IN_ + k] : Whh0[(size_t)n * H_ + (k - IN_)];
        Wcat0[i] = (bf16_t)v;
    }
    // layer1 input == layer1 hidden  =>  pre-sum the two weight matrices
    for (size_t i = tid; i < (size_t)G4_ * H_; i += nth)
        W1s[i] = (bf16_t)(Wih1[i] + Whh1[i]);
    for (size_t i = tid; i < (size_t)IN_ * H_; i += nth)
        Wlinb[i] = (bf16_t)Wlin[i];
    for (size_t i = tid; i < (size_t)G4_; i += nth) {
        bias0[i] = bih0[i] + bhh0[i];
        bias1[i] = bih1[i] + bhh1[i];
    }
    // xh = [x0 | h0] in bf16
    for (size_t i = tid; i < (size_t)B_ * KC0; i += nth) {
        int b = (int)(i / KC0), k = (int)(i % KC0);
        float v = (k < IN_) ? x0[(size_t)b * IN_ + k] : h0[(size_t)b * H_ + (k - IN_)];
        xh[i] = (bf16_t)v;
    }
    for (size_t i = tid; i < (size_t)B_ * H_; i += nth) cbuf[i] = c0[i];
    for (size_t i = tid; i < (size_t)NBAR * NWG; i += nth) arrive[i] = 0u;
}

// ---------------- persistent cooperative LSTM kernel ----------------
// Work decomposition per gate GEMM (64 x 6144):
//   96 j-tiles of 16 columns over H; WG owns j-tiles {2wg, 2wg+1}.
//   Per j-tile: 4 M-tiles x 4 gates = 16 output tiles; a wave register-blocks
//   2 M-tiles x 2 gates (4 accumulators), so 4 waves cover one j-tile and
//   8 waves cover the WG's two j-tiles. 2 A-frags + 2 B-frags -> 4 WMMAs/K-step.
__global__ void __launch_bounds__(NTHR, 1) lstm_persistent_kernel(
    const float* __restrict__ blin, float* __restrict__ out,
    const bf16_t* __restrict__ Wcat0, const bf16_t* __restrict__ W1s,
    const bf16_t* __restrict__ Wlinb,
    const float* __restrict__ bias0, const float* __restrict__ bias1,
    bf16_t* xh, bf16_t* h1b, float* gates, float* cbuf, unsigned int* arrive)
{
    const int tid  = threadIdx.x;
    const int wg   = blockIdx.x;
    const int lane = tid & 31;
    const int wave = tid >> 5;            // 8 waves
    const int jt   = wg * 2 + (wave >> 2);     // this wave's j-tile (0..95)
    const int j0   = jt * 16;
    const int v4   = wave & 3;            // position within the j-tile's 4 waves
    const int m0   = (v4 & 1) * 32;       // M-tiles m0, m0+16
    const int gp   = (v4 >> 1) * 2;       // gates gp, gp+1
    const int jcol0 = wg * 32;            // WG's 32 elementwise columns
    const int nl   = lane & 15;
    const int mh   = (lane >> 4) * 8;

    for (int t = 0; t < T_; ++t) {
        // ===== Phase A: gates0 = [x|h] @ Wcat0^T  (K = 1664) =====
        {
            const int nA = gp * H_ + j0;
            const int nB = (gp + 1) * H_ + j0;
            v8f acc00 = {}, acc01 = {}, acc10 = {}, acc11 = {};
            #pragma unroll 2
            for (int k0 = 0; k0 < KC0; k0 += 32) {
                v16bf a0 = load_a_frag(xh, KC0, m0,      k0, lane);
                v16bf a1 = load_a_frag(xh, KC0, m0 + 16, k0, lane);
                v16bf b0 = load_b_frag(Wcat0, KC0, nA, k0, lane);
                v16bf b1 = load_b_frag(Wcat0, KC0, nB, k0, lane);
                acc00 = wmma_bf16(a0, b0, acc00);
                acc01 = wmma_bf16(a0, b1, acc01);
                acc10 = wmma_bf16(a1, b0, acc10);
                acc11 = wmma_bf16(a1, b1, acc11);
            }
            #pragma unroll
            for (int v = 0; v < 8; ++v) {
                int ma = m0 + mh + v, mb = m0 + 16 + mh + v;
                gates[(size_t)ma * G4_ + nA + nl] = acc00[v];
                gates[(size_t)ma * G4_ + nB + nl] = acc01[v];
                gates[(size_t)mb * G4_ + nA + nl] = acc10[v];
                gates[(size_t)mb * G4_ + nB + nl] = acc11[v];
            }
        }
        __syncthreads();
        // cell update (layer 0) — WG-local: this WG produced all 4 gates for its 32 cols
        for (int e = tid; e < B_ * 32; e += NTHR) {
            int b = e >> 5, j = jcol0 + (e & 31);
            size_t r = (size_t)b * G4_;
            float gi = gates[r + j]          + bias0[j];
            float gf = gates[r + H_ + j]     + bias0[H_ + j];
            float gg = gates[r + 2 * H_ + j] + bias0[2 * H_ + j];
            float go = gates[r + 3 * H_ + j] + bias0[3 * H_ + j];
            float c1 = sigmoidf_(gf) * cbuf[(size_t)b * H_ + j] + sigmoidf_(gi) * tanhf(gg);
            float h1 = sigmoidf_(go) * tanhf(c1);
            cbuf[(size_t)b * H_ + j] = c1;
            h1b[(size_t)b * H_ + j] = (bf16_t)h1;
        }
        grid_barrier(arrive, 3 * t + 0, wg, tid);

        // ===== Phase B: gates1 = h1 @ (Wih1+Whh1)^T  (K = 1536) =====
        {
            const int nA = gp * H_ + j0;
            const int nB = (gp + 1) * H_ + j0;
            v8f acc00 = {}, acc01 = {}, acc10 = {}, acc11 = {};
            #pragma unroll 2
            for (int k0 = 0; k0 < H_; k0 += 32) {
                v16bf a0 = load_a_frag(h1b, H_, m0,      k0, lane);
                v16bf a1 = load_a_frag(h1b, H_, m0 + 16, k0, lane);
                v16bf b0 = load_b_frag(W1s, H_, nA, k0, lane);
                v16bf b1 = load_b_frag(W1s, H_, nB, k0, lane);
                acc00 = wmma_bf16(a0, b0, acc00);
                acc01 = wmma_bf16(a0, b1, acc01);
                acc10 = wmma_bf16(a1, b0, acc10);
                acc11 = wmma_bf16(a1, b1, acc11);
            }
            #pragma unroll
            for (int v = 0; v < 8; ++v) {
                int ma = m0 + mh + v, mb = m0 + 16 + mh + v;
                gates[(size_t)ma * G4_ + nA + nl] = acc00[v];
                gates[(size_t)ma * G4_ + nB + nl] = acc01[v];
                gates[(size_t)mb * G4_ + nA + nl] = acc10[v];
                gates[(size_t)mb * G4_ + nB + nl] = acc11[v];
            }
        }
        __syncthreads();
        // cell update (layer 1): h2 goes (bf16) straight into the h-slot of xh for next step
        for (int e = tid; e < B_ * 32; e += NTHR) {
            int b = e >> 5, j = jcol0 + (e & 31);
            size_t r = (size_t)b * G4_;
            float gi = gates[r + j]          + bias1[j];
            float gf = gates[r + H_ + j]     + bias1[H_ + j];
            float gg = gates[r + 2 * H_ + j] + bias1[2 * H_ + j];
            float go = gates[r + 3 * H_ + j] + bias1[3 * H_ + j];
            float c2 = sigmoidf_(gf) * cbuf[(size_t)b * H_ + j] + sigmoidf_(gi) * tanhf(gg);
            float h2 = sigmoidf_(go) * tanhf(c2);
            cbuf[(size_t)b * H_ + j] = c2;
            xh[(size_t)b * KC0 + IN_ + j] = (bf16_t)h2;
            if (t == T_ - 1) {   // final hidden-state outputs, torch layout (1,B,H)
                out[(size_t)B_ * T_ * IN_ + (size_t)b * H_ + j] = h2;
                out[(size_t)B_ * T_ * IN_ + (size_t)B_ * H_ + (size_t)b * H_ + j] = c2;
            }
        }
        grid_barrier(arrive, 3 * t + 1, wg, tid);

        // ===== Phase C: out_t = h2 @ Wlin^T + blin  (64x128, K = 1536) =====
        if (wg < 32 && wave == 0) {
            const int mm0 = (wg >> 3) * 16;
            const int nn0 = (wg & 7) * 16;
            v8f acc = {};
            #pragma unroll 2
            for (int k0 = 0; k0 < H_; k0 += 32) {
                v16bf a = load_a_frag(xh, KC0, mm0, IN_ + k0, lane);   // h-part of xh
                v16bf b = load_b_frag(Wlinb, H_, nn0, k0, lane);
                acc = wmma_bf16(a, b, acc);
            }
            int cn = nn0 + nl;
            float bl = blin[cn];
            #pragma unroll
            for (int v = 0; v < 8; ++v) {
                int m = mm0 + mh + v;
                float val = acc[v] + bl;
                out[((size_t)m * T_ + t) * IN_ + cn] = val;  // outputs (B,T,IN)
                xh[(size_t)m * KC0 + cn] = (bf16_t)val;      // next step's x (bf16)
            }
        }
        if (t != T_ - 1) grid_barrier(arrive, 3 * t + 2, wg, tid);
    }
}

// ---------------- launch ----------------
extern "C" void kernel_launch(void* const* d_in, const int* in_sizes, int n_in,
                              void* d_out, int out_size, void* d_ws, size_t ws_size,
                              hipStream_t stream) {
    const float* x0   = (const float*)d_in[0];
    const float* h0   = (const float*)d_in[1];
    const float* c0   = (const float*)d_in[2];
    const float* Wih0 = (const float*)d_in[3];
    const float* Whh0 = (const float*)d_in[4];
    const float* bih0 = (const float*)d_in[5];
    const float* bhh0 = (const float*)d_in[6];
    const float* Wih1 = (const float*)d_in[7];
    const float* Whh1 = (const float*)d_in[8];
    const float* bih1 = (const float*)d_in[9];
    const float* bhh1 = (const float*)d_in[10];
    const float* Wlin = (const float*)d_in[11];
    const float* blin = (const float*)d_in[12];
    // d_in[13] = target_len (fixed T_=32 at compile time)

    char* ws = (char*)d_ws;
    bf16_t* Wcat0 = (bf16_t*)(ws + OFF_WCAT0);
    bf16_t* W1s   = (bf16_t*)(ws + OFF_W1S);
    bf16_t* Wlinb = (bf16_t*)(ws + OFF_WLIN);
    float*  bias0 = (float*)(ws + OFF_BIAS0);
    float*  bias1 = (float*)(ws + OFF_BIAS1);
    bf16_t* xh    = (bf16_t*)(ws + OFF_XH);
    bf16_t* h1b   = (bf16_t*)(ws + OFF_H1B);
    float*  gates = (float*)(ws + OFF_GATES);
    float*  cbuf  = (float*)(ws + OFF_CBUF);
    unsigned int* arrive = (unsigned int*)(ws + OFF_ARRIVE);

    lstm_init_kernel<<<1024, 256, 0, stream>>>(
        Wih0, Whh0, bih0, bhh0, Wih1, Whh1, bih1, bhh1, Wlin,
        x0, h0, c0, Wcat0, W1s, Wlinb, bias0, bias1, xh, cbuf, arrive);

    lstm_persistent_kernel<<<NWG, NTHR, 0, stream>>>(
        blin, (float*)d_out, Wcat0, W1s, Wlinb, bias0, bias1,
        xh, h1b, gates, cbuf, arrive);
}
